// Multi_class_34574486733597
// MI455X (gfx1250) — matrix operand order repaired and checked by reading, not compiled, not run on hardware
//
#include <hip/hip_runtime.h>
#include <hip/hip_bf16.h>

// ---------------------------------------------------------------------------
// MulT-style 9-stream x 2-layer cross-modal transformer forward for MI455X.
// Compute-bound (~1.26 TFLOP vs ~508MB weight traffic -> ~2.5 kFLOP/B), so all
// GEMMs run through v_wmma_f32_16x16x32_f16 (fp32 global -> f16 LDS tiles,
// f32 accumulate). Wave32; fragment layouts per CDNA5 ISA 7.12.2.
// Main GEMM: double-buffered LDS software pipeline (1 barrier / k-step).
// Async-to-LDS (ASYNCcnt) used for the f16 attention-prob tiles.
// ---------------------------------------------------------------------------

typedef __attribute__((ext_vector_type(16))) _Float16 v16h;
typedef __attribute__((ext_vector_type(8)))  float    v8f;

#define HEADS 12
#define DH 64
#define DIMM 768

// ---- WMMA fragment builders (CDNA5 ISA 7.12.2, wave32) --------------------
__device__ inline v16h frag_a(const _Float16* base, int stride, int m0, int kk, int lane) {
  int m = lane & 15, h = lane >> 4;
  union { v16h v; unsigned u[8]; } f;
  const _Float16* rp = base + (size_t)(m0 + m) * stride + kk;
#pragma unroll
  for (int j = 0; j < 4; ++j)
    f.u[j] = *(const unsigned*)(rp + h * 8 + 2 * j);
#pragma unroll
  for (int j = 0; j < 4; ++j)
    f.u[4 + j] = *(const unsigned*)(rp + 16 + h * 8 + 2 * j);
  return f.v;
}

// B fragment 32x16 f16 (Kdim x N); LDS holds B transposed: T[n][k].
__device__ inline v16h frag_b(const _Float16* baseT, int stride, int n0, int kk, int lane) {
  int n = lane & 15, h = lane >> 4;
  union { v16h v; unsigned u[8]; } f;
  const _Float16* rp = baseT + (size_t)(n0 + n) * stride + kk;
#pragma unroll
  for (int j = 0; j < 4; ++j) f.u[j] = *(const unsigned*)(rp + h * 8 + 2 * j);
#pragma unroll
  for (int j = 0; j < 4; ++j) f.u[4 + j] = *(const unsigned*)(rp + 16 + h * 8 + 2 * j);
  return f.v;
}

// ---- Generic GEMM: C[M,N] = act(A[M,K] @ B[K,N] + bias + pos) --------------
#define GF_BIAS 1
#define GF_RELU 2
#define GF_POS  4

__global__ __launch_bounds__(256)
void gemm_wmma_f16(const float* __restrict__ A, const float* __restrict__ B,
                   const float* __restrict__ bias, const float* __restrict__ pos,
                   float* __restrict__ C,
                   int M, int N, int K, int lda, int ldb, int ldc,
                   int Ltok, int flags) {
  __shared__ _Float16 As[2][128 * 40];
  __shared__ _Float16 Bs[2][64 * 40];   // transposed: Bs[n][k]
  const int tid = threadIdx.x, lane = tid & 31, wave = tid >> 5;
  const int wm = (wave >> 1) * 32, wn = (wave & 1) * 32;
  const int mT = blockIdx.y * 128, nT = blockIdx.x * 64;

  const v8f vzero = {0.f, 0.f, 0.f, 0.f, 0.f, 0.f, 0.f, 0.f};
  v8f acc[2][2];
  acc[0][0] = vzero; acc[0][1] = vzero; acc[1][0] = vzero; acc[1][1] = vzero;

  const int arow = tid >> 1, acol = (tid & 1) * 16;   // A: 128x32, 16 per thread
  const int kp = tid >> 4, nb = (tid & 15) * 4;       // B: 2 k-rows x 4 n per thread

  const int kFull = K & ~31;

  union ARegs { _Float16 h[16]; uint4 q[2]; } ar;
  union H2 { _Float16 h[2]; unsigned u; };
  unsigned br[4];

  // fetch tile k0 from global into registers (f32 -> f16 convert)
  auto fetch = [&](int k0) {
    const float* ap = A + (size_t)(mT + arow) * lda + k0 + acol;
    if (k0 + 32 < kFull) __builtin_prefetch(ap + 32, 0, 1);   // global_prefetch_b8
#pragma unroll
    for (int i = 0; i < 4; ++i) {
      float4 va = *(const float4*)(ap + 4 * i);
      ar.h[4 * i + 0] = (_Float16)va.x; ar.h[4 * i + 1] = (_Float16)va.y;
      ar.h[4 * i + 2] = (_Float16)va.z; ar.h[4 * i + 3] = (_Float16)va.w;
    }
    const float* bp0 = B + (size_t)(k0 + 2 * kp) * ldb + nT + nb;
    float4 vb0 = *(const float4*)bp0;
    float4 vb1 = *(const float4*)(bp0 + ldb);
    H2 p;
    p.h[0] = (_Float16)vb0.x; p.h[1] = (_Float16)vb1.x; br[0] = p.u;
    p.h[0] = (_Float16)vb0.y; p.h[1] = (_Float16)vb1.y; br[1] = p.u;
    p.h[0] = (_Float16)vb0.z; p.h[1] = (_Float16)vb1.z; br[2] = p.u;
    p.h[0] = (_Float16)vb0.w; p.h[1] = (_Float16)vb1.w; br[3] = p.u;
  };
  // stage registers into LDS buffer
  auto stage = [&](int buf) {
    *(uint4*)(As[buf] + arow * 40 + acol) = ar.q[0];
    *(uint4*)(As[buf] + arow * 40 + acol + 8) = ar.q[1];
#pragma unroll
    for (int j = 0; j < 4; ++j)
      *(unsigned*)(Bs[buf] + (nb + j) * 40 + 2 * kp) = br[j];
  };
  auto mma = [&](int buf) {
#pragma unroll
    for (int am = 0; am < 2; ++am) {
      v16h af = frag_a(As[buf], 40, wm + am * 16, 0, lane);
#pragma unroll
      for (int bn = 0; bn < 2; ++bn) {
        v16h bf = frag_b(Bs[buf], 40, wn + bn * 16, 0, lane);
        acc[am][bn] = __builtin_amdgcn_wmma_f32_16x16x32_f16(
            false, af, false, bf, (short)0, acc[am][bn], false, false);
      }
    }
  };

  int cur = 0;
  if (kFull > 0) {                 // software pipeline over full 32-wide tiles
    fetch(0); stage(0);
    __syncthreads();
    for (int k0 = 32; k0 < kFull; k0 += 32) {
      fetch(k0);                   // global loads overlap with...
      mma(cur);                    // ...WMMA on the staged tile
      stage(cur ^ 1);
      __syncthreads();
      cur ^= 1;
    }
  }

  if (kFull < K) {                 // guarded tail (audio conv K=40 only)
    {
      const float* ap = A + (size_t)(mT + arow) * lda + kFull + acol;
      _Float16* dst = As[cur ^ 1] + arow * 40 + acol;
#pragma unroll
      for (int i = 0; i < 16; ++i)
        dst[i] = (_Float16)((kFull + acol + i < K) ? ap[i] : 0.f);
    }
    {
      int k0r = kFull + 2 * kp, k1r = k0r + 1;
#pragma unroll
      for (int j = 0; j < 4; ++j) {
        float f0 = (k0r < K) ? B[(size_t)k0r * ldb + nT + nb + j] : 0.f;
        float f1 = (k1r < K) ? B[(size_t)k1r * ldb + nT + nb + j] : 0.f;
        H2 p; p.h[0] = (_Float16)f0; p.h[1] = (_Float16)f1;
        *(unsigned*)(Bs[cur ^ 1] + (nb + j) * 40 + 2 * kp) = p.u;
      }
    }
    if (kFull > 0) mma(cur);       // last pipelined tile overlaps tail staging
    __syncthreads();
    mma(cur ^ 1);
  } else if (kFull > 0) {
    mma(cur);                      // drain pipeline
  }

  const int hh = lane >> 4, nn = lane & 15;
#pragma unroll
  for (int am = 0; am < 2; ++am)
#pragma unroll
    for (int bn = 0; bn < 2; ++bn) {
      int gn = nT + wn + bn * 16 + nn;
      float bv = (flags & GF_BIAS) ? bias[gn] : 0.f;
      union { v8f v; float f[8]; } u; u.v = acc[am][bn];
#pragma unroll
      for (int r = 0; r < 8; ++r) {
        int gm = mT + wm + am * 16 + hh * 8 + r;
        float v = u.f[r] + bv;
        if (flags & GF_POS) v += pos[(size_t)(gm % Ltok) * DIMM + gn];
        if (flags & GF_RELU) v = fmaxf(v, 0.f);
        C[(size_t)gm * ldc + gn] = v;
      }
    }
}

// ---- Attention scores: S[b,h,q,k] = (Q.Kt)/8, per-(b,h) 64x64 tiles --------
__global__ __launch_bounds__(256)
void attn_qk(const float* __restrict__ Q, const float* __restrict__ Kb,
             float* __restrict__ S, int Lq, int Lk) {
  __shared__ _Float16 Qs[64 * 72];
  __shared__ _Float16 Ks[64 * 72];
  const int bh = blockIdx.z, b = bh / HEADS, h = bh % HEADS;
  const int qT = blockIdx.y * 64, kT = blockIdx.x * 64;
  const int tid = threadIdx.x, lane = tid & 31, wave = tid >> 5;
  const int wm = (wave >> 1) * 16, wn = (wave & 1) * 32;

  const int row = tid >> 2, colB = (tid & 3) * 16;   // 64x64, 16 per thread
  {
    const float* qp = Q + ((size_t)(b * Lq + qT + row) * DIMM + h * DH + colB);
    const float* kp = Kb + ((size_t)(b * Lk + kT + row) * DIMM + h * DH + colB);
    union { _Float16 h[16]; uint4 q[2]; } tq, tk;
#pragma unroll
    for (int i = 0; i < 4; ++i) {
      float4 a = *(const float4*)(qp + 4 * i);
      float4 c = *(const float4*)(kp + 4 * i);
      tq.h[4 * i + 0] = (_Float16)a.x; tq.h[4 * i + 1] = (_Float16)a.y;
      tq.h[4 * i + 2] = (_Float16)a.z; tq.h[4 * i + 3] = (_Float16)a.w;
      tk.h[4 * i + 0] = (_Float16)c.x; tk.h[4 * i + 1] = (_Float16)c.y;
      tk.h[4 * i + 2] = (_Float16)c.z; tk.h[4 * i + 3] = (_Float16)c.w;
    }
    *(uint4*)(Qs + row * 72 + colB) = tq.q[0];
    *(uint4*)(Qs + row * 72 + colB + 8) = tq.q[1];
    *(uint4*)(Ks + row * 72 + colB) = tk.q[0];
    *(uint4*)(Ks + row * 72 + colB + 8) = tk.q[1];
  }
  __syncthreads();

  const v8f vzero = {0.f, 0.f, 0.f, 0.f, 0.f, 0.f, 0.f, 0.f};
  v8f acc[2]; acc[0] = vzero; acc[1] = vzero;
#pragma unroll
  for (int kk = 0; kk < 64; kk += 32) {
    v16h af = frag_a(Qs, 72, wm, kk, lane);
#pragma unroll
    for (int bn = 0; bn < 2; ++bn) {
      v16h bf = frag_b(Ks, 72, wn + bn * 16, kk, lane);  // Ks[n][d]: pairs along d
      acc[bn] = __builtin_amdgcn_wmma_f32_16x16x32_f16(
          false, af, false, bf, (short)0, acc[bn], false, false);
    }
  }
  const int hh = lane >> 4, nn = lane & 15;
#pragma unroll
  for (int bn = 0; bn < 2; ++bn) {
    union { v8f v; float f[8]; } u; u.v = acc[bn];
#pragma unroll
    for (int r = 0; r < 8; ++r) {
      int gq = qT + wm + hh * 8 + r, gk = kT + wn + bn * 16 + nn;
      S[((size_t)bh * Lq + gq) * Lk + gk] = u.f[r] * 0.125f;   // 1/sqrt(64)
    }
  }
}

// ---- Fused: s = s + c*s_prev - 1e8*(1-mask); write-back; softmax -> f16 ----
__global__ __launch_bounds__(256)
void fuse_softmax(float* __restrict__ S, const float* __restrict__ Sprev,
                  const float* __restrict__ cptr, const float* __restrict__ mask,
                  _Float16* __restrict__ P, int Lk, int rowsPerBatch, int totalRows) {
  const int wave = threadIdx.x >> 5, lane = threadIdx.x & 31;
  const int row = blockIdx.x * 8 + wave;
  if (row >= totalRows) return;
  const int b = row / rowsPerBatch;
  const float c = (Sprev != nullptr) ? cptr[0] : 0.f;
  const int nk = Lk >> 5;                 // Lk in {64,128,256} -> <= 8
  float r[8];
  float mx = -3.4e38f;
#pragma unroll
  for (int i = 0; i < 8; ++i)
    if (i < nk) {
      int k = i * 32 + lane;
      size_t idx = (size_t)row * Lk + k;
      float s = S[idx];
      if (Sprev) s += c * Sprev[idx];
      s -= 1e8f * (1.f - mask[b * Lk + k]);
      S[idx] = s;                          // fused+masked scores_prev for next layer
      r[i] = s; mx = fmaxf(mx, s);
    }
#pragma unroll
  for (int o = 16; o; o >>= 1) mx = fmaxf(mx, __shfl_xor(mx, o, 32));
  float sum = 0.f;
#pragma unroll
  for (int i = 0; i < 8; ++i)
    if (i < nk) { r[i] = __expf(r[i] - mx); sum += r[i]; }
#pragma unroll
  for (int o = 16; o; o >>= 1) sum += __shfl_xor(sum, o, 32);
  const float inv = 1.f / sum;
#pragma unroll
  for (int i = 0; i < 8; ++i)
    if (i < nk) P[(size_t)row * Lk + i * 32 + lane] = (_Float16)(r[i] * inv);
}

// ---- O[b,q,h*64+d] = sum_k P[b,h,q,k] * V[b,k,h*64+d] ----------------------
// P tiles are f16 already -> pure copy: CDNA5 async-to-LDS (ASYNCcnt).
__global__ __launch_bounds__(256)
void attn_av(const _Float16* __restrict__ P, const float* __restrict__ V,
             float* __restrict__ O, int Lq, int Lk) {
  __shared__ _Float16 Ps[64 * 40];
  __shared__ _Float16 Vs[64 * 40];   // transposed: Vs[d][k]
  const int bh = blockIdx.z, b = bh / HEADS, h = bh % HEADS;
  const int qT = blockIdx.y * 64;
  const int tid = threadIdx.x, lane = tid & 31, wave = tid >> 5;
  const int wm = (wave >> 1) * 16, wn = (wave & 1) * 32;

  const int prow = tid >> 2, pcol = (tid & 3) * 8;   // P: 64x32 f16, 16B per thread
  const int vkp = tid >> 4, vdb = (tid & 15) * 4;    // V: 2 k-rows x 4 d per thread

  // LDS byte offset of this thread's P-tile slot (generic ptr low 32 bits).
  const unsigned psOff = (unsigned)(size_t)(&Ps[prow * 40 + pcol]);

  const v8f vzero = {0.f, 0.f, 0.f, 0.f, 0.f, 0.f, 0.f, 0.f};
  v8f acc[2]; acc[0] = vzero; acc[1] = vzero;

  union H2 { _Float16 h[2]; unsigned u; };

  for (int k0 = 0; k0 < Lk; k0 += 32) {
    {  // async b128 copy global->LDS, one 16B transfer per lane
      const _Float16* gsrc = P + ((size_t)bh * Lq + qT + prow) * Lk + k0 + pcol;
      asm volatile("global_load_async_to_lds_b128 %0, %1, off"
                   :: "v"(psOff), "v"(gsrc) : "memory");
    }
    {  // V tile: k-pair packed transposed stores (4x ds_store_b32)
      const float* vp0 = V + ((size_t)(b * Lk + k0 + 2 * vkp) * DIMM + h * DH + vdb);
      float4 v0 = *(const float4*)vp0;
      float4 v1 = *(const float4*)(vp0 + DIMM);
      H2 p;
      p.h[0] = (_Float16)v0.x; p.h[1] = (_Float16)v1.x;
      *(unsigned*)(Vs + (vdb + 0) * 40 + 2 * vkp) = p.u;
      p.h[0] = (_Float16)v0.y; p.h[1] = (_Float16)v1.y;
      *(unsigned*)(Vs + (vdb + 1) * 40 + 2 * vkp) = p.u;
      p.h[0] = (_Float16)v0.z; p.h[1] = (_Float16)v1.z;
      *(unsigned*)(Vs + (vdb + 2) * 40 + 2 * vkp) = p.u;
      p.h[0] = (_Float16)v0.w; p.h[1] = (_Float16)v1.w;
      *(unsigned*)(Vs + (vdb + 3) * 40 + 2 * vkp) = p.u;
    }
    asm volatile("s_wait_asynccnt 0x0" ::: "memory");
    __syncthreads();
    v16h af = frag_a(Ps, 40, wm, 0, lane);
#pragma unroll
    for (int bn = 0; bn < 2; ++bn) {
      v16h bf = frag_b(Vs, 40, wn + bn * 16, 0, lane);
      acc[bn] = __builtin_amdgcn_wmma_f32_16x16x32_f16(
          false, af, false, bf, (short)0, acc[bn], false, false);
    }
    __syncthreads();
  }
  const int hh = lane >> 4, nn = lane & 15;
#pragma unroll
  for (int bn = 0; bn < 2; ++bn) {
    union { v8f v; float f[8]; } u; u.v = acc[bn];
#pragma unroll
    for (int r = 0; r < 8; ++r) {
      int gq = qT + wm + hh * 8 + r, gd = wn + bn * 16 + nn;
      O[(size_t)(b * Lq + gq) * DIMM + h * DH + gd] = u.f[r];
    }
  }
}

// ---- Out = LayerNorm(X + s*R) * g + bt ; one wave per 768-wide row ---------
__global__ __launch_bounds__(256)
void resln(const float* __restrict__ X, const float* __restrict__ R,
           const float* __restrict__ sp, const float* __restrict__ g,
           const float* __restrict__ bt, float* __restrict__ Out, int rows) {
  const int wave = threadIdx.x >> 5, lane = threadIdx.x & 31;
  const int row = blockIdx.x * 8 + wave;
  if (row >= rows) return;
  const float s = sp[0];
  const float* x = X + (size_t)row * DIMM;
  const float* rr = R + (size_t)row * DIMM;
  float v[24];
  float sum = 0.f;
#pragma unroll
  for (int i = 0; i < 24; ++i) {
    int idx = i * 32 + lane;
    float t = x[idx] + s * rr[idx];
    v[i] = t; sum += t;
  }
#pragma unroll
  for (int o = 16; o; o >>= 1) sum += __shfl_xor(sum, o, 32);
  const float mean = sum * (1.f / 768.f);
  float var = 0.f;
#pragma unroll
  for (int i = 0; i < 24; ++i) { float d = v[i] - mean; var += d * d; }
#pragma unroll
  for (int o = 16; o; o >>= 1) var += __shfl_xor(var, o, 32);
  const float rstd = rsqrtf(var * (1.f / 768.f) + 1e-5f);
  float* op = Out + (size_t)row * DIMM;
#pragma unroll
  for (int i = 0; i < 24; ++i) {
    int idx = i * 32 + lane;
    op[idx] = (v[i] - mean) * rstd * g[idx] + bt[idx];
  }
}

// ---- qout buffer offsets (floats) -----------------------------------------
__host__ __device__ inline size_t qoff(int g, int i) {
  const size_t szL = 32ull * 128 * DIMM, szV = 32ull * 64 * DIMM, szA = 32ull * 256 * DIMM;
  if (g < 3) return (size_t)(g * 2 + i) * szL;
  if (g < 6) return 6 * szL + (size_t)((g - 3) * 2 + i) * szV;
  return 6 * szL + 6 * szV + (size_t)((g - 6) * 2 + i) * szA;
}

// ---- mean/max pool over 448 concatenated tokens ---------------------------
__global__ void pool_kernel(const float* __restrict__ qout, float* __restrict__ pooled) {
  int t = blockIdx.x * blockDim.x + threadIdx.x;
  if (t >= 32 * 4608) return;
  const int b = t / 4608, f = t % 4608;
  const int j = f / DIMM, d = f % DIMM, s = j >> 1, i = j & 1;
  const float* Lb = qout + qoff(s, i)     + (size_t)b * 128 * DIMM + d;
  const float* Ab = qout + qoff(6 + s, i) + (size_t)b * 256 * DIMM + d;
  const float* Vb = qout + qoff(3 + s, i) + (size_t)b * 64 * DIMM + d;
  float sum = 0.f, mx = -3.4e38f;
  for (int k = 0; k < 128; ++k) { float x = Lb[(size_t)k * DIMM]; sum += x; mx = fmaxf(mx, x); }
  for (int k = 0; k < 256; ++k) { float x = Ab[(size_t)k * DIMM]; sum += x; mx = fmaxf(mx, x); }
  for (int k = 0; k < 64;  ++k) { float x = Vb[(size_t)k * DIMM]; sum += x; mx = fmaxf(mx, x); }
  pooled[(size_t)b * 9216 + f] = sum * (1.f / 448.f);
  pooled[(size_t)b * 9216 + 4608 + f] = mx;
}

// ---- classifier: out[b,c] = pooled[b,:] . W[:,c] + bias[c] ----------------
__global__ __launch_bounds__(256)
void cls_kernel(const float* __restrict__ pooled, const float* __restrict__ W,
                const float* __restrict__ bias, float* __restrict__ out) {
  const int wave = threadIdx.x >> 5, lane = threadIdx.x & 31;
  const int idx = blockIdx.x * 8 + wave;
  if (idx >= 32 * 7) return;
  const int b = idx / 7, c = idx % 7;
  const float* p = pooled + (size_t)b * 9216;
  float sum = 0.f;
  for (int f = lane; f < 9216; f += 32) sum += p[f] * W[(size_t)f * 7 + c];
#pragma unroll
  for (int o = 16; o; o >>= 1) sum += __shfl_xor(sum, o, 32);
  if (lane == 0) out[b * 7 + c] = sum + bias[c];
}

// ---------------------------------------------------------------------------
// Host orchestration.
// Input flattening assumption (insertion order of the setup_inputs dicts):
//  0:l 1:v_256 2:v_512 3:v_1024 4:a 5:l_mask 6:v_mask 7:a_mask
//  8..17: conv wl,bl,w256,b256,w512,b512,w1024,b1024,wa,ba
//  18..287: 18 blocks x {wq,wk,wv,wp,g1,b1,g2,b2,w1,bf1,w2,bf2,a,b,c}
//  288:pos_l 289:pos_v 290:pos_a 291:cls_w 292:cls_b
// ---------------------------------------------------------------------------
extern "C" void kernel_launch(void* const* d_in, const int* in_sizes, int n_in,
                              void* d_out, int out_size, void* d_ws, size_t ws_size,
                              hipStream_t stream) {
  (void)in_sizes; (void)n_in; (void)out_size; (void)ws_size;
  const float* inL    = (const float*)d_in[0];
  const float* inV256 = (const float*)d_in[1];
  const float* inV512 = (const float*)d_in[2];
  const float* inV1024= (const float*)d_in[3];
  const float* inA    = (const float*)d_in[4];
  const float* maskL  = (const float*)d_in[5];
  const float* maskV  = (const float*)d_in[6];
  const float* maskA  = (const float*)d_in[7];
  const float* pos_l  = (const float*)d_in[288];
  const float* pos_v  = (const float*)d_in[289];
  const float* pos_a  = (const float*)d_in[290];
  const float* cls_w  = (const float*)d_in[291];
  const float* cls_b  = (const float*)d_in[292];
  float* out = (float*)d_out;

  // workspace carve-up
  char* ws = (char*)d_ws;
  size_t cur = 0;
  auto allocB = [&](size_t bytes) -> char* {
    char* p = ws + cur; cur = (cur + bytes + 255) & ~(size_t)255; return p;
  };
  auto allocF = [&](size_t elems) -> float* { return (float*)allocB(elems * 4); };

  const size_t SZL = 32ull * 128 * DIMM, SZV = 32ull * 64 * DIMM, SZA = 32ull * 256 * DIMM;
  float* lu   = allocF(SZL);
  float* vu   = allocF(SZV);
  float* au   = allocF(SZA);
  float* qout = allocF(6 * (SZL + SZV + SZA));
  float* bufQ  = allocF(SZA);
  float* bufK  = allocF(SZA);
  float* bufV  = allocF(SZA);
  float* bufO  = allocF(SZA);
  float* bufO2 = allocF(SZA);
  float* bufQ1 = allocF(SZA);
  float* bufF  = allocF(32ull * 256 * 3072);
  const size_t SZS = 32ull * HEADS * 256 * 256;
  float* S0 = allocF(SZS);
  float* S1 = allocF(SZS);
  _Float16* Pb = (_Float16*)allocB(SZS * 2);
  float* pooled = allocF(32ull * 9216);

  auto gemm = [&](const float* A, const float* Bw, const float* bias, const float* pos,
                  float* C, int M, int N, int K, int lda, int ldb, int ldc,
                  int Ltok, int flags) {
    dim3 grid(N / 64, M / 128);
    gemm_wmma_f16<<<grid, 256, 0, stream>>>(A, Bw, bias, pos, C, M, N, K,
                                            lda, ldb, ldc, Ltok, flags);
  };

  // ---- input projections (conv1d k=1) + positional embeddings ----
  gemm(inL, (const float*)d_in[8], (const float*)d_in[9], pos_l,
       lu, 32 * 128, 768, 768, 768, 768, 768, 128, GF_BIAS | GF_POS);
  gemm(inV256, (const float*)d_in[10], (const float*)d_in[11], pos_v,
       vu, 32 * 64, 256, 256, 256, 256, 768, 64, GF_BIAS | GF_POS);
  gemm(inV512, (const float*)d_in[12], (const float*)d_in[13], pos_v + 256,
       vu + 256, 32 * 64, 256, 512, 512, 256, 768, 64, GF_BIAS | GF_POS);
  gemm(inV1024, (const float*)d_in[14], (const float*)d_in[15], pos_v + 512,
       vu + 512, 32 * 64, 256, 1024, 1024, 256, 768, 64, GF_BIAS | GF_POS);
  gemm(inA, (const float*)d_in[16], (const float*)d_in[17], pos_a,
       au, 32 * 256, 768, 40, 40, 768, 768, 256, GF_BIAS | GF_POS);

  // ---- 9 streams x 2 layers ----
  const float* srcQ[9] = {lu, lu, lu, vu, vu, vu, au, au, au};
  const float* srcKV[9] = {lu, vu, au, vu, lu, au, au, lu, vu};
  const float* msk[9] = {maskL, maskV, maskA, maskV, maskL, maskA, maskA, maskL, maskV};
  const int LQ[9] = {128, 128, 128, 64, 64, 64, 256, 256, 256};
  const int LK[9] = {128, 64, 256, 64, 128, 256, 256, 128, 64};

  for (int g = 0; g < 9; ++g) {
    const int Lq = LQ[g], Lk = LK[g];
    const int Mq = 32 * Lq, Mk = 32 * Lk;
    const float* kv = srcKV[g];
    for (int i = 0; i < 2; ++i) {
      const int bb = 18 + 15 * (g * 2 + i);
      auto bw = [&](int k) { return (const float*)d_in[bb + k]; };
      const float* qin = (i == 0) ? srcQ[g] : qout + qoff(g, 0);
      float* qo = qout + qoff(g, i);
      float* Scur = (i == 0) ? S0 : S1;
      const float* Sprev = (i == 0) ? nullptr : S0;

      gemm(qin, bw(0), nullptr, nullptr, bufQ, Mq, 768, 768, 768, 768, 768, 0, 0);
      gemm(kv,  bw(1), nullptr, nullptr, bufK, Mk, 768, 768, 768, 768, 768, 0, 0);
      gemm(kv,  bw(2), nullptr, nullptr, bufV, Mk, 768, 768, 768, 768, 768, 0, 0);

      attn_qk<<<dim3(Lk / 64, Lq / 64, 32 * HEADS), 256, 0, stream>>>(bufQ, bufK, Scur, Lq, Lk);

      const int totalRows = 32 * HEADS * Lq;
      fuse_softmax<<<(totalRows + 7) / 8, 256, 0, stream>>>(
          Scur, Sprev, bw(14), msk[g], Pb, Lk, HEADS * Lq, totalRows);

      attn_av<<<dim3(1, Lq / 64, 32 * HEADS), 256, 0, stream>>>(Pb, bufV, bufO, Lq, Lk);

      gemm(bufO, bw(3), nullptr, nullptr, bufO2, Mq, 768, 768, 768, 768, 768, 0, 0);

      resln<<<(Mq + 7) / 8, 256, 0, stream>>>(qin, bufO2, bw(12), bw(4), bw(5), bufQ1, Mq);

      gemm(bufQ1, bw(8), bw(9), nullptr, bufF, Mq, 3072, 768, 768, 3072, 3072, 0,
           GF_BIAS | GF_RELU);
      gemm(bufF, bw(10), bw(11), nullptr, bufO, Mq, 768, 3072, 3072, 768, 768, 0, GF_BIAS);

      resln<<<(Mq + 7) / 8, 256, 0, stream>>>(bufQ1, bufO, bw(13), bw(6), bw(7), qo, Mq);
    }
  }

  // ---- pooling + classifier ----
  pool_kernel<<<(32 * 4608 + 255) / 256, 256, 0, stream>>>(qout, pooled);
  cls_kernel<<<(32 * 7 + 7) / 8, 256, 0, stream>>>(pooled, cls_w, cls_b, out);
}
